// QuantizedCNN_66821101191757
// MI455X (gfx1250) — matrix operand order repaired and checked by reading, not compiled, not run on hardware
//
#include <hip/hip_runtime.h>

typedef __attribute__((ext_vector_type(16))) __bf16 v16bf;
typedef __attribute__((ext_vector_type(8)))  __bf16 v8bf;
typedef __attribute__((ext_vector_type(8)))  float  v8f;
typedef __attribute__((ext_vector_type(4)))  int    v4i;

// ---------------------------------------------------------------------------
// Workspace layout (bytes):
//   [0, 16384)                         : fc1 f32 accumulator [32][128]
//   [16384, 16384+33554432)            : h1  bf16 NHWC [32][128][128][32]
//   [+33554432, +33554432+16777216)    : h2  bf16 [32][262144]  (NCHW flatten)
// ---------------------------------------------------------------------------
#define OFF_H1 16384
#define OFF_H2 (16384 + 33554432)

// --- CDNA5 async global->LDS copy (ASYNCcnt path), builtin or inline asm ----
__device__ __forceinline__ void async_copy16(void* lds_dst, const void* gsrc) {
#if __has_builtin(__builtin_amdgcn_global_load_async_to_lds_b128)
    __builtin_amdgcn_global_load_async_to_lds_b128(
        (__attribute__((address_space(1))) v4i*)gsrc,
        (__attribute__((address_space(3))) v4i*)lds_dst, 0, 0);
#else
    unsigned lo = (unsigned)(unsigned long long)(__attribute__((address_space(3))) void*)lds_dst;
    unsigned long long ga = (unsigned long long)gsrc;
    asm volatile("global_load_async_to_lds_b128 %0, %1, off" :: "v"(lo), "v"(ga) : "memory");
#endif
}

__device__ __forceinline__ void wait_async0() {
#if __has_builtin(__builtin_amdgcn_s_wait_asynccnt)
    __builtin_amdgcn_s_wait_asynccnt(0);
#else
    asm volatile("s_wait_asynccnt 0x0" ::: "memory");
#endif
}

__global__ void k_zero_acc(float* __restrict__ acc) {
    acc[blockIdx.x * 256 + threadIdx.x] = 0.0f;   // 16 blocks * 256 = 4096
}

// ---------------------------------------------------------------------------
// conv1 (1->32, 3x3, pad 1) + bias + ReLU + maxpool2 -> h1 NHWC bf16
// One thread per pooled pixel (b,y,x); computes all 32 channels. 1.2 GFLOP.
// ---------------------------------------------------------------------------
__global__ void k_conv1(const float* __restrict__ x, const float* __restrict__ w1,
                        const float* __restrict__ b1, unsigned short* __restrict__ h1u) {
    __shared__ float Wl[288];
    __shared__ float Bl[32];
    int tid = threadIdx.x;
    if (tid < 288) Wl[tid] = w1[tid];
    if (tid < 32)  Bl[tid] = b1[tid];
    __syncthreads();

    __bf16* h1 = (__bf16*)h1u;
    int idx = blockIdx.x * 256 + tid;          // 524288 total
    int b   = idx >> 14;
    int rem = idx & 16383;
    int y   = rem >> 7;
    int xp  = rem & 127;
    const float* in = x + (size_t)b * 65536;

    float v[4][4];
    int iy0 = 2 * y - 1, ix0 = 2 * xp - 1;
#pragma unroll
    for (int r = 0; r < 4; ++r) {
        int iy = iy0 + r;
#pragma unroll
        for (int c = 0; c < 4; ++c) {
            int ix = ix0 + c;
            bool ok = (iy >= 0) && (iy < 256) && (ix >= 0) && (ix < 256);
            v[r][c] = ok ? in[iy * 256 + ix] : 0.0f;
        }
    }

    __bf16 outv[32];
    for (int ch = 0; ch < 32; ++ch) {
        float s00 = 0.f, s01 = 0.f, s10 = 0.f, s11 = 0.f;
#pragma unroll
        for (int ky = 0; ky < 3; ++ky)
#pragma unroll
            for (int kx = 0; kx < 3; ++kx) {
                float w = Wl[ch * 9 + ky * 3 + kx];
                s00 += v[ky][kx] * w;     s01 += v[ky][kx + 1] * w;
                s10 += v[ky + 1][kx] * w; s11 += v[ky + 1][kx + 1] * w;
            }
        float m = fmaxf(fmaxf(s00, s01), fmaxf(s10, s11)) + Bl[ch];
        outv[ch] = (__bf16)fmaxf(m, 0.0f);
    }
    __bf16* dst = h1 + (size_t)idx * 32;
#pragma unroll
    for (int i = 0; i < 4; ++i) ((v8bf*)dst)[i] = ((v8bf*)outv)[i];
}

// ---------------------------------------------------------------------------
// conv2 as implicit GEMM on WMMA bf16 + bias + ReLU + fused maxpool2.
// Grid: 4096 = 32(b) * 64(pooled row Y) * 2(x-half). Block 256 = 8 waves.
// Input tile (4 rows x 66 px x 32 ch, halo included) staged into LDS via
// async global->LDS b128 copies; halo zero-filled with disjoint ds stores.
// GEMM per wave: M=16 pixels, N=64 channels (4 tiles), K=288 (9 taps * 32 ch).
// ---------------------------------------------------------------------------
__global__ void k_conv2(const unsigned short* __restrict__ h1u,
                        const float* __restrict__ w2, const float* __restrict__ b2,
                        unsigned short* __restrict__ h2u) {
    __shared__ __bf16 Wl[64 * 288];          // 36864 B, [o][r*32 + c], r = kh*3+kw
    __shared__ __bf16 inTile[4 * 68 * 32];   // 17408 B, [row4][x(68; 66 used)][ch32]
    __shared__ __bf16 stage[2 * 64 * 64];    // 16384 B, [row(2)][x(64)][ch(64)]
    const __bf16* h1 = (const __bf16*)h1u;
    __bf16* h2 = (__bf16*)h2u;
    int tid  = threadIdx.x;
    int gid  = blockIdx.x;
    int b    = gid >> 7;
    int Y    = (gid >> 1) & 63;
    int half = gid & 1;

    // weight transform w2[o][c][r] (OIHW flat) -> Wl[o][r*32+c], fp32->bf16
    for (int i = tid; i < 64 * 288; i += 256) {
        int o = i / 288, rm = i % 288, c = rm / 9, r = rm % 9;
        Wl[o * 288 + r * 32 + c] = (__bf16)w2[i];
    }

    // stage input rows 2Y-1..2Y+2, x = half*64-1 .. half*64+64 (66 px) in 16B chunks
    for (int ci = tid; ci < 4 * 66 * 4; ci += 256) {
        int q   = ci & 3;                    // 16B quarter of a 64B pixel vector
        int pl  = ci >> 2;
        int r4  = pl / 66;
        int xl  = pl - r4 * 66;
        int yy2 = 2 * Y - 1 + r4;
        int xx  = half * 64 - 1 + xl;
        __bf16* lp = &inTile[(r4 * 68 + xl) * 32 + q * 8];
        if ((yy2 >= 0) && (yy2 < 128) && (xx >= 0) && (xx < 128)) {
            const __bf16* gp = h1 + ((((size_t)b * 128 + yy2) * 128 + xx) << 5) + q * 8;
            async_copy16(lp, gp);
        } else {
            v8bf z;
#pragma unroll
            for (int i = 0; i < 8; ++i) z[i] = (__bf16)0.0f;
            *(v8bf*)lp = z;                  // zero halo (disjoint from async writes)
        }
    }
    wait_async0();
    __syncthreads();

    int w  = tid >> 5, lane = tid & 31;
    int ln = lane & 15;
    int hi = lane >> 4;                      // which K sub-block this lane holds
    int rowl  = w & 1;                       // pre-pool row within the Y pair
    int xbase = (w >> 1) * 16;               // 16-pixel chunk within the 64-px half
    int cofs  = hi * 16;                     // A: els0-15 = ch cofs..+15

    v8f acc[4];
#pragma unroll
    for (int t = 0; t < 4; ++t)
#pragma unroll
        for (int j = 0; j < 8; ++j) acc[t][j] = 0.0f;

    for (int r = 0; r < 9; ++r) {
        int dy = r / 3, dx = r % 3;          // 0..2 ; tile row0 = 2Y-1, tile x0 = half*64-1
        int r4 = rowl + dy;
        int xl = xbase + ln + dx;
        const v8bf* src = (const v8bf*)&inTile[(r4 * 68 + xl) * 32 + cofs];
        v8bf lo = src[0], hv = src[1];
        v16bf a;
#pragma unroll
        for (int i = 0; i < 8; ++i) { a[i] = lo[i]; a[i + 8] = hv[i]; }
#pragma unroll
        for (int nt = 0; nt < 4; ++nt) {
            int o = nt * 16 + ln;            // B: lane = N, els = K hi*16 .. hi*16+15
            const v8bf* wsrc = (const v8bf*)(&Wl[o * 288 + r * 32 + hi * 16]);
            v8bf w0 = wsrc[0], w1v = wsrc[1];
            v16bf bm;
#pragma unroll
            for (int i = 0; i < 8; ++i) { bm[i] = w0[i]; bm[i + 8] = w1v[i]; }
            acc[nt] = __builtin_amdgcn_wmma_f32_16x16x32_bf16(
                false, a, false, bm, (short)0, acc[nt], false, false);
        }
    }

    // bias + ReLU, stage pre-pool tile. C layout: VGPR j -> M=j (lanes<16), 8+j (lanes>=16)
#pragma unroll
    for (int nt = 0; nt < 4; ++nt) {
        int ch = nt * 16 + ln;
        float bias = b2[ch];
#pragma unroll
        for (int j = 0; j < 8; ++j) {
            int px = xbase + j + hi * 8;
            float v = fmaxf(acc[nt][j] + bias, 0.0f);
            stage[(rowl * 64 + px) * 64 + ch] = (__bf16)v;
        }
    }
    __syncthreads();

    // 2x2 maxpool + store to h2 in NCHW-flattened [b][ch*4096 + Y*64 + X]
    int ch  = tid >> 2;
    int pxb = (tid & 3) * 8;
    v8bf outv;
#pragma unroll
    for (int i = 0; i < 8; ++i) {
        int px = pxb + i;
        float a0 = (float)stage[(2 * px) * 64 + ch];
        float a1 = (float)stage[(2 * px + 1) * 64 + ch];
        float a2 = (float)stage[(64 + 2 * px) * 64 + ch];
        float a3 = (float)stage[(64 + 2 * px + 1) * 64 + ch];
        outv[i] = (__bf16)fmaxf(fmaxf(a0, a1), fmaxf(a2, a3));
    }
    size_t off = (size_t)b * 262144 + (size_t)ch * 4096 + (size_t)Y * 64 + half * 32 + pxb;
    *(v8bf*)(h2 + off) = outv;
}

// ---------------------------------------------------------------------------
// fc1: [32 x 262144] @ dequant-int4 [262144 x 128] via WMMA bf16, split-K.
// 512 blocks, each handles a K-chunk of 512 (16 steps of 32) for all M,N.
// HBM-bound on the 134 MB code stream -> prefetch 8 K-steps ahead.
// ---------------------------------------------------------------------------
__global__ void k_fc1(const unsigned short* __restrict__ h2u,
                      const int* __restrict__ codes, const float* __restrict__ scales,
                      float* __restrict__ acc) {
    const __bf16* h2 = (const __bf16*)h2u;
    int tid = threadIdx.x;
    int w = tid >> 5, lane = tid & 31, ln = lane & 15, hi = lane >> 4;
    int n0 = w * 16;
    int n  = n0 + ln;
    int kbase = blockIdx.x * 512;

    v8f c0, c1;
#pragma unroll
    for (int j = 0; j < 8; ++j) { c0[j] = 0.0f; c1[j] = 0.0f; }

    const int* crow = codes + (size_t)n * 262144;
    int kofsB = hi * 16;                   // B: contiguous K chunk of 16
    int kofsA = hi * 16;                   // A: K kofsA..+15

    for (int s = 0; s < 16; ++s) {
        int k  = kbase + s * 32;
        int kb = k + kofsB;
        __builtin_prefetch(crow + kb + 256, 0, 0);   // global_prefetch_b8, 8 steps ahead
        // dequant 16 codes -> bf16 B operand (one scale per 16-aligned chunk, BLOCK=64)
        float scale = scales[((size_t)n * 262144 + (size_t)kb) >> 6];
        v16bf bm;
#pragma unroll
        for (int i = 0; i < 16; ++i) bm[i] = (__bf16)((float)crow[kb + i] * scale);

        // A operands, batches 0-15 and 16-31
        const v8bf* a0lo = (const v8bf*)(h2 + (size_t)ln * 262144 + k + kofsA);
        const v8bf* a0hi = (const v8bf*)(h2 + (size_t)ln * 262144 + k + kofsA + 8);
        const v8bf* a1lo = (const v8bf*)(h2 + (size_t)(16 + ln) * 262144 + k + kofsA);
        const v8bf* a1hi = (const v8bf*)(h2 + (size_t)(16 + ln) * 262144 + k + kofsA + 8);
        v8bf l0 = a0lo[0], h0 = a0hi[0], l1 = a1lo[0], h1v = a1hi[0];
        v16bf a0, a1;
#pragma unroll
        for (int i = 0; i < 8; ++i) {
            a0[i] = l0[i]; a0[i + 8] = h0[i];
            a1[i] = l1[i]; a1[i + 8] = h1v[i];
        }
        c0 = __builtin_amdgcn_wmma_f32_16x16x32_bf16(false, a0, false, bm, (short)0, c0, false, false);
        c1 = __builtin_amdgcn_wmma_f32_16x16x32_bf16(false, a1, false, bm, (short)0, c1, false, false);
    }

#pragma unroll
    for (int j = 0; j < 8; ++j) {
        int mb = j + hi * 8;
        atomicAdd(&acc[mb * 128 + n0 + ln], c0[j]);
        atomicAdd(&acc[(mb + 16) * 128 + n0 + ln], c1[j]);
    }
}

// ---------------------------------------------------------------------------
// epilogue: h3 = relu(acc + fc1_bias);  out = h3 @ dequant(W2).T + fc2_bias
// ---------------------------------------------------------------------------
__global__ void k_fc2(const float* __restrict__ acc, const float* __restrict__ fc1b,
                      const int* __restrict__ c2, const float* __restrict__ s2,
                      const float* __restrict__ b2, float* __restrict__ out) {
    int t = threadIdx.x;                   // 64 threads
    int b = t >> 1, j = t & 1;
    float sum = b2[j];
    for (int ncol = 0; ncol < 128; ++ncol) {
        float h  = fmaxf(acc[b * 128 + ncol] + fc1b[ncol], 0.0f);
        float wv = (float)c2[j * 128 + ncol] * s2[(j * 128 + ncol) >> 6];
        sum += h * wv;
    }
    out[b * 2 + j] = sum;
}

extern "C" void kernel_launch(void* const* d_in, const int* in_sizes, int n_in,
                              void* d_out, int out_size, void* d_ws, size_t ws_size,
                              hipStream_t stream) {
    const float* x          = (const float*)d_in[0];
    const float* w1         = (const float*)d_in[1];
    const float* b1         = (const float*)d_in[2];
    const float* w2         = (const float*)d_in[3];
    const float* b2         = (const float*)d_in[4];
    const int*   fc1_codes  = (const int*)d_in[5];
    const float* fc1_scales = (const float*)d_in[6];
    const float* fc1_bias   = (const float*)d_in[7];
    const int*   fc2_codes  = (const int*)d_in[8];
    const float* fc2_scales = (const float*)d_in[9];
    const float* fc2_bias   = (const float*)d_in[10];

    char* ws = (char*)d_ws;
    float*          acc = (float*)ws;
    unsigned short* h1  = (unsigned short*)(ws + OFF_H1);
    unsigned short* h2  = (unsigned short*)(ws + OFF_H2);

    k_zero_acc<<<16, 256, 0, stream>>>(acc);
    k_conv1<<<2048, 256, 0, stream>>>(x, w1, b1, h1);
    k_conv2<<<4096, 256, 0, stream>>>(h1, w2, b2, h2);
    k_fc1<<<512, 256, 0, stream>>>(h2, fc1_codes, fc1_scales, acc);
    k_fc2<<<1, 64, 0, stream>>>(acc, fc1_bias, fc2_codes, fc2_scales, fc2_bias, (float*)d_out);
}
